// GatedDeltaProductAdapter_85237920957010
// MI455X (gfx1250) — compile-verified, matrix operand order checked by reading, not run on hardware
//
#include <hip/hip_runtime.h>
#include <hip/hip_bf16.h>

// Problem constants (match reference)
#define BB   2
#define TT   2048
#define DD   256
#define NTOK (BB*TT)   // 4096 tokens

typedef _Float16 v16h __attribute__((ext_vector_type(16)));
typedef _Float16 v8h  __attribute__((ext_vector_type(8)));
typedef float    v8f  __attribute__((ext_vector_type(8)));

__device__ __forceinline__ float silu_f(float a){ return a / (1.0f + __expf(-a)); }

// ---------------- CDNA5 async global->LDS copy (ASYNCcnt path), inline asm ----
__device__ __forceinline__ void async_ld_b128(void* lds_dst, const void* gsrc){
  uint32_t lds_off = (uint32_t)(uintptr_t)lds_dst;   // low 32 bits of generic = LDS byte offset
  asm volatile("global_load_async_to_lds_b128 %0, %1, off"
               :: "v"(lds_off), "v"(gsrc) : "memory");
}
__device__ __forceinline__ void async_ld_b32(void* lds_dst, const void* gsrc){
  uint32_t lds_off = (uint32_t)(uintptr_t)lds_dst;
  asm volatile("global_load_async_to_lds_b32 %0, %1, off"
               :: "v"(lds_off), "v"(gsrc) : "memory");
}
__device__ __forceinline__ void wait_async0(){
  asm volatile("s_wait_asynccnt 0x0" ::: "memory");
}

// ---------------------------------------------------------------- cast x -> f16
__global__ void k_cast_x(const float* __restrict__ x, _Float16* __restrict__ xh, int n){
  int i = blockIdx.x*blockDim.x + threadIdx.x;
  if (i < n) xh[i] = (_Float16)x[i];
}

// ------------------------------------------- transpose+cast 7 weight matrices
// WT layout: WT[mat][n][k] = W[k][n], mats: Wq, Wk0, Wk1, Wv0, Wv1, Wg, Wo
__global__ void k_prep_w(const float* __restrict__ Wq, const float* __restrict__ Wk,
                         const float* __restrict__ Wv, const float* __restrict__ Wg,
                         const float* __restrict__ Wo, _Float16* __restrict__ WT){
  int i = blockIdx.x*blockDim.x + threadIdx.x;
  if (i >= 7*65536) return;
  int mat = i >> 16;
  int nn  = (i >> 8) & 255;
  int kk  = i & 255;
  const float* src;
  switch (mat){
    case 0:  src = Wq;          break;
    case 1:  src = Wk;          break;
    case 2:  src = Wk + 65536;  break;
    case 3:  src = Wv;          break;
    case 4:  src = Wv + 65536;  break;
    case 5:  src = Wg;          break;
    default: src = Wo;          break;
  }
  WT[i] = (_Float16)src[kk*256 + nn];
}

// -------------------------------------------------------- WMMA f16 GEMM
// C[z][M=4096, N=256] = A[4096,256] * W_z (W supplied transposed, f16).
// One wave = one 16x64 tile (4 accumulators): A fragment reused across 4 WMMAs.
__global__ void k_gemm(const _Float16* __restrict__ A, const _Float16* __restrict__ BtBase,
                       float* __restrict__ CBase){
  const int mt   = blockIdx.x;                 // 0..255
  const int wave = threadIdx.x >> 5;           // 0..3
  const int nt0  = wave*4;                     // 4 n-tiles (64 cols) per wave
  const _Float16* Bt = BtBase + (size_t)blockIdx.z*65536;
  float* C = CBase + (size_t)blockIdx.z*((size_t)NTOK*256);
  const int lane = threadIdx.x & 31;
  const int row  = lane & 15;
  const int hi   = lane >> 4;
  const size_t aRow = (size_t)(mt*16 + row)*256;
  size_t bRow[4];
  #pragma unroll
  for (int i=0;i<4;i++) bRow[i] = (size_t)((nt0+i)*16 + row)*256;

  v8f acc[4];
  #pragma unroll
  for (int i=0;i<4;i++)
    #pragma unroll
    for (int r=0;r<8;r++) acc[i][r] = 0.0f;

  #pragma unroll
  for (int kb=0; kb<256; kb+=32){
    // 16-bit A/B WMMA VGPR layout: lane row holds K = kb + hi*8 + {0..7} and
    // K = kb + 16 + hi*8 + {0..7}  -> two contiguous 16B loads per operand.
    v8h a0 = *(const v8h*)(A + aRow + kb + hi*8);
    v8h a1 = *(const v8h*)(A + aRow + kb + 16 + hi*8);
    v16h av;
    #pragma unroll
    for (int i=0;i<8;i++){ av[i]=a0[i]; av[i+8]=a1[i]; }
    #pragma unroll
    for (int nb=0; nb<4; nb++){
      v8h b0 = *(const v8h*)(Bt + bRow[nb] + kb + hi*8);
      v8h b1 = *(const v8h*)(Bt + bRow[nb] + kb + 16 + hi*8);
      v16h bv;
      #pragma unroll
      for (int i=0;i<8;i++){ bv[i]=b0[i]; bv[i+8]=b1[i]; }
      acc[nb] = __builtin_amdgcn_wmma_f32_16x16x32_f16(
          /*neg_a=*/false, av, /*neg_b=*/false, bv,
          /*c_mod=*/(short)0, acc[nb], /*reuse_a=*/false, /*reuse_b=*/false);
    }
  }

  #pragma unroll
  for (int nb=0; nb<4; nb++)
    #pragma unroll
    for (int r=0;r<8;r++)
      C[(size_t)(mt*16 + r + hi*8)*256 + (nt0+nb)*16 + row] = acc[nb][r];
}

// ---------------------------------------------------------------- reductions
__device__ __forceinline__ float block_sum_256(float v, float* red){
  const int c = threadIdx.x;
  red[c] = v; __syncthreads();
  #pragma unroll
  for (int s=128; s>0; s>>=1){ if (c < s) red[c] += red[c+s]; __syncthreads(); }
  float r = red[0]; __syncthreads();
  return r;
}

// ----------------- per-token: causal conv4 + SiLU, L2 norm, beta, gate-decay g
__global__ void k_token_prep(const float* __restrict__ x,
                             const float* __restrict__ pre,   // 6 slabs: q,k0,k1,v0,v1,g
                             const float* __restrict__ qcw, const float* __restrict__ kcw,
                             const float* __restrict__ vcw,
                             const float* __restrict__ Wb, const float* __restrict__ Wa,
                             const float* __restrict__ A_log, const float* __restrict__ dt_bias,
                             float* __restrict__ post,        // 5 slabs: q,k0,k1,v0,v1
                             float* __restrict__ beta, float* __restrict__ g)
{
  __shared__ float red[256];
  const int bt = blockIdx.x;
  const int c  = threadIdx.x;
  const int b  = bt / TT, t = bt % TT;
  const size_t slab = (size_t)NTOK*256;

  auto conv_silu = [&](const float* p, const float* w)->float{
    float a = 0.0f;
    #pragma unroll
    for (int i=0;i<4;i++){
      int tt = t - 3 + i;
      if (tt >= 0) a += p[(size_t)(b*TT + tt)*256 + c] * w[c*4 + i];
    }
    return silu_f(a);
  };

  float xq  = conv_silu(pre + 0*slab, qcw);
  float xk0 = conv_silu(pre + 1*slab, kcw);
  float xk1 = conv_silu(pre + 2*slab, kcw + 1024);
  float xv0 = conv_silu(pre + 3*slab, vcw);
  float xv1 = conv_silu(pre + 4*slab, vcw + 1024);
  float xc  = x[(size_t)bt*256 + c];

  float sq  = block_sum_256(xq*xq,   red);
  float sk0 = block_sum_256(xk0*xk0, red);
  float sk1 = block_sum_256(xk1*xk1, red);
  float db0 = block_sum_256(xc*Wb[c],       red);
  float db1 = block_sum_256(xc*Wb[256 + c], red);
  float da  = block_sum_256(xc*Wa[c],       red);

  const size_t idx = (size_t)bt*256 + c;
  post[0*slab + idx] = xq  * rsqrtf(sq  + 1e-6f);
  post[1*slab + idx] = xk0 * rsqrtf(sk0 + 1e-6f);
  post[2*slab + idx] = xk1 * rsqrtf(sk1 + 1e-6f);
  post[3*slab + idx] = xv0;
  post[4*slab + idx] = xv1;
  if (c == 0){
    beta[bt*2 + 0] = 1.0f / (1.0f + __expf(-db0));
    beta[bt*2 + 1] = 1.0f / (1.0f + __expf(-db1));
    float aval = da + dt_bias[0];
    float sp = (aval > 20.0f) ? aval : log1pf(__expf(aval));
    g[bt] = -__expf(A_log[0]) * sp;
  }
}

// ------------- sequential gated delta scan: state S (256x256 f32) lives in LDS
// One workgroup per batch; 1024 threads; ~278 KB dynamic LDS (fits 320 KB/WGP).
// Per-step inputs (q,k0,k1,v0,v1,g,beta) are double-buffered in LDS and
// prefetched one step ahead with global_load_async_to_lds (ASYNCcnt), so the
// steady-state loop never blocks on global load latency.
#define VBUF 1296   // floats per buffer: 5*256 vectors + 3 scalars, padded to 16B
__global__ void k_scan(const float* __restrict__ post, const float* __restrict__ beta,
                       const float* __restrict__ g, float* __restrict__ o)
{
  extern __shared__ float smem[];
  float* S    = smem;              // 65536 : S[d][v], row-major
  float* vecs = S + 65536;         // 2*VBUF: double-buffered q|k0|k1|v0|v1|scal
  float* u    = vecs + 2*VBUF;     // 256
  float* part = u + 256;           // 1024
  const int b = blockIdx.x, tid = threadIdx.x;
  const size_t slab = (size_t)NTOK*256;

  // issue async prefetch of step `t`'s inputs into buffer `buf`
  auto issue_step = [&](int t, float* buf){
    const size_t bt = (size_t)b*TT + t;
    if (tid < 320){
      int vI = tid >> 6, e4 = (tid & 63)*4;
      async_ld_b128(buf + vI*256 + e4, post + (size_t)vI*slab + bt*256 + e4);
    } else if (tid == 320){
      async_ld_b32(buf + 1280, g + bt);
    } else if (tid == 321){
      async_ld_b32(buf + 1281, beta + 2*bt);
    } else if (tid == 322){
      async_ld_b32(buf + 1282, beta + 2*bt + 1);
    }
  };

  for (int e=0;e<64;e++) S[e*1024 + tid] = 0.0f;

  issue_step(0, vecs);          // prologue prefetch
  wait_async0();
  __syncthreads();

  for (int t=0;t<TT;t++){
    const size_t bt = (size_t)b*TT + t;
    float* bufc = vecs + (t & 1)*VBUF;
    float* bufn = vecs + ((t + 1) & 1)*VBUF;

    // overlap: kick off next step's vectors while this step computes
    if (t + 1 < TT) issue_step(t + 1, bufn);

    const float eg = __expf(bufc[1280]);
    for (int e=0;e<64;e++) S[e*1024 + tid] *= eg;    // S *= exp(g_t)
    __syncthreads();

    #pragma unroll
    for (int j=0;j<2;j++){
      const float* kv = bufc + 256 + j*256;
      const float* vv = bufc + 768 + j*256;
      const float bj = bufc[1281 + j];
      const int v = tid & 255, p = tid >> 8;
      // kS[v] = sum_d k[d]*S[d][v]  (4 partial threads per column)
      float acc = 0.0f;
      for (int d = p*64; d < p*64+64; d++) acc += kv[d]*S[d*256 + v];
      part[tid] = acc;
      __syncthreads();
      if (tid < 256)
        u[tid] = bj * (vv[tid] - (part[tid]+part[tid+256]+part[tid+512]+part[tid+768]));
      __syncthreads();
      // S += k (x) u
      for (int e=0;e<64;e++){
        int i = e*1024 + tid;
        S[i] += kv[i>>8] * u[i & 255];
      }
      __syncthreads();
    }

    { // o[v] = sum_d q[d]*S[d][v]
      const int v = tid & 255, p = tid >> 8;
      float acc = 0.0f;
      for (int d = p*64; d < p*64+64; d++) acc += bufc[d]*S[d*256 + v];
      part[tid] = acc;
      __syncthreads();
      if (tid < 256)
        o[bt*256 + tid] = part[tid]+part[tid+256]+part[tid+512]+part[tid+768];
    }

    wait_async0();              // next-step buffer resident before anyone reads it
    __syncthreads();
  }
}

// ---------------------------- RMS-norm * o_norm_w * SiLU(gate), emit f16 for Wo
__global__ void k_gate(const float* __restrict__ o, const float* __restrict__ gpre,
                       const float* __restrict__ onw, _Float16* __restrict__ og){
  __shared__ float red[256];
  const int bt = blockIdx.x, c = threadIdx.x;
  const size_t idx = (size_t)bt*256 + c;
  float ov = o[idx];
  float ms = block_sum_256(ov*ov, red) * (1.0f/256.0f);
  float gt = gpre[idx];
  og[idx] = (_Float16)(ov * rsqrtf(ms + 1e-5f) * onw[c] * silu_f(gt));
}

// ================================================================ launch
extern "C" void kernel_launch(void* const* d_in, const int* in_sizes, int n_in,
                              void* d_out, int out_size, void* d_ws, size_t ws_size,
                              hipStream_t stream) {
  (void)in_sizes; (void)n_in; (void)out_size; (void)ws_size;
  const float* x     = (const float*)d_in[0];
  const float* Wq    = (const float*)d_in[1];
  const float* Wk    = (const float*)d_in[2];
  const float* Wv    = (const float*)d_in[3];
  const float* Wb    = (const float*)d_in[4];
  const float* Wa    = (const float*)d_in[5];
  const float* A_log = (const float*)d_in[6];
  const float* dtb   = (const float*)d_in[7];
  const float* qcw   = (const float*)d_in[8];
  const float* kcw   = (const float*)d_in[9];
  const float* vcw   = (const float*)d_in[10];
  const float* Wg    = (const float*)d_in[11];
  const float* onw   = (const float*)d_in[12];
  const float* Wo    = (const float*)d_in[13];
  float* out = (float*)d_out;

  char* ws = (char*)d_ws;
  size_t off = 0;
  _Float16* xh = (_Float16*)(ws+off); off += (size_t)NTOK*256*sizeof(_Float16);
  _Float16* WT = (_Float16*)(ws+off); off += (size_t)7*65536*sizeof(_Float16);
  float* pre   = (float*)(ws+off);    off += (size_t)6*NTOK*256*sizeof(float);
  float* post  = (float*)(ws+off);    off += (size_t)5*NTOK*256*sizeof(float);
  float* beta  = (float*)(ws+off);    off += (size_t)NTOK*2*sizeof(float);
  float* gdec  = (float*)(ws+off);    off += (size_t)NTOK*sizeof(float);
  float* ob    = (float*)(ws+off);    off += (size_t)NTOK*256*sizeof(float);
  _Float16* og = (_Float16*)(ws+off); off += (size_t)NTOK*256*sizeof(_Float16);

  // 1) operand prep (f16 cast + weight transpose)
  k_cast_x<<<(NTOK*256+255)/256, 256, 0, stream>>>(x, xh, NTOK*256);
  k_prep_w<<<(7*65536+255)/256, 256, 0, stream>>>(Wq, Wk, Wv, Wg, Wo, WT);

  // 2) six projection GEMMs via WMMA (z = Wq,Wk0,Wk1,Wv0,Wv1,Wg)
  k_gemm<<<dim3(NTOK/16, 1, 6), 128, 0, stream>>>(xh, WT, pre);

  // 3) conv/SiLU/L2-norm + beta + decay per token
  k_token_prep<<<NTOK, 256, 0, stream>>>(x, pre, qcw, kcw, vcw, Wb, Wa, A_log, dtb,
                                         post, beta, gdec);

  // 4) sequential scan: S resident in LDS, inputs async-prefetched to LDS
  const size_t scanLds = (size_t)(65536 + 2*VBUF + 256 + 1024)*sizeof(float);
  k_scan<<<BB, 1024, scanLds, stream>>>(post, beta, gdec, ob);

  // 5) output gating (RMS norm * SiLU gate), then final WMMA GEMM with Wo
  k_gate<<<NTOK, 256, 0, stream>>>(ob, pre + (size_t)5*NTOK*256, onw, og);
  k_gemm<<<dim3(NTOK/16, 1, 1), 128, 0, stream>>>(og, WT + (size_t)6*65536, out);
}
